// Attention_16569983828124
// MI455X (gfx1250) — matrix operand order repaired and checked by reading, not compiled, not run on hardware
//
#include <hip/hip_runtime.h>

typedef __attribute__((ext_vector_type(16))) __bf16 v16bf;
typedef __attribute__((ext_vector_type(8)))  float  v8f;
typedef unsigned short ushort_t;
typedef unsigned int   uint_t;

#define B_   4
#define SQ_  2048
#define SK_  2048
#define D_   1024
#define INV_SCALE 0.03125f   /* 1/sqrt(1024) */
#define LDS_STRIDE 40        /* 32 bf16 + pad; 80B rows -> conflict-free 16B reads */

union Frag16 { v16bf v; uint4 u[2]; };

// Pack two fp32 into two bf16 (round-half-up) with a single v_perm_b32.
// dst[15:0] = hi16(a + 0x8000), dst[31:16] = hi16(b + 0x8000)
__device__ __forceinline__ uint_t pack_bf2(float a, float b) {
    uint_t ua = __float_as_uint(a) + 0x8000u;
    uint_t ub = __float_as_uint(b) + 0x8000u;
    // selector bytes: dst0=src1.b2, dst1=src1.b3, dst2=src0.b2, dst3=src0.b3
    return __builtin_amdgcn_perm(ub, ua, 0x07060302u);
}

__device__ __forceinline__ ushort_t f2bf1(float f) {
    return (ushort_t)((__float_as_uint(f) + 0x8000u) >> 16);
}

__device__ __forceinline__ void store_bf4(ushort_t* p, float4 f) {
    uint2 w;
    w.x = pack_bf2(f.x, f.y);
    w.y = pack_bf2(f.z, f.w);
    *(uint2*)p = w;
}

// A-fragment: 16x32 bf16. lane = 16*half + m. Per lane: k = half*8..+7 and 16+half*8..+7.
__device__ __forceinline__ void load_fragA(Frag16& f, const ushort_t* lds, int row, int lane) {
    int m    = lane & 15;
    int half = lane >> 4;
    int off  = (row + m) * LDS_STRIDE + half * 8;
    f.u[0] = *(const uint4*)(lds + off);
    f.u[1] = *(const uint4*)(lds + off + 16);
}

// B-fragment: 32x16 bf16. lane = 16*kg + n. Per lane: k = kg*8..+7 and 16+kg*8..+7 at "column" n.
// LDS tile stored [n][k] (k contiguous) -> 2 contiguous 16B reads.
__device__ __forceinline__ void load_fragB(Frag16& f, const ushort_t* lds, int row, int lane) {
    int n  = lane & 15;
    int kg = lane >> 4;
    int off = (row + n) * LDS_STRIDE + kg * 8;
    f.u[0] = *(const uint4*)(lds + off);
    f.u[1] = *(const uint4*)(lds + off + 16);
}

// ---------------- Kernel 1: S = mask(Q @ K^T) * inv_scale ----------------
// grid = (SK/128, SQ/128, B), block = 256 (8 waves, 4x2 wave grid, 32x64 per wave)
__global__ __launch_bounds__(256) void qk_kernel(const float* __restrict__ Q,
                                                 const float* __restrict__ K,
                                                 float* __restrict__ S) {
    const int kt = blockIdx.x, qt = blockIdx.y, b = blockIdx.z;
    if (kt > qt) return;                       // causal tile skip (uniform)
    const int q0 = qt * 128, k0 = kt * 128;

    __shared__ ushort_t ldsQ[2][128 * LDS_STRIDE];
    __shared__ ushort_t ldsK[2][128 * LDS_STRIDE];

    const int tid = threadIdx.x, lane = tid & 31, wid = tid >> 5;
    const int wm = (wid >> 1) * 32;            // wave M origin (0..96)
    const int wn = (wid & 1) * 64;             // wave N origin (0 or 64)

    const int frow = tid >> 3;                 // fill row base (this thread)
    const int fcol = (tid & 7) * 4;            // fill col

    v8f acc[2][4];
    for (int i = 0; i < 2; ++i)
        for (int j = 0; j < 4; ++j)
            acc[i][j] = {};

    const size_t qbase = ((size_t)b * SQ_ + q0) * D_;
    const size_t kbase = ((size_t)b * SK_ + k0) * D_;

    float4 pq[4], pk[4];
    // prefetch k-step 0
    for (int i = 0; i < 4; ++i) {
        int row = frow + 32 * i;
        pq[i] = *(const float4*)(Q + qbase + (size_t)row * D_ + fcol);
        pk[i] = *(const float4*)(K + kbase + (size_t)row * D_ + fcol);
    }
    for (int i = 0; i < 4; ++i) {
        int row = frow + 32 * i;
        store_bf4(&ldsQ[0][row * LDS_STRIDE + fcol], pq[i]);
        store_bf4(&ldsK[0][row * LDS_STRIDE + fcol], pk[i]);
    }
    __syncthreads();

    for (int ks = 0; ks < D_ / 32; ++ks) {
        const int cur = ks & 1;
        if (ks + 1 < D_ / 32) {                // prefetch next tile into registers
            for (int i = 0; i < 4; ++i) {
                int row = frow + 32 * i;
                pq[i] = *(const float4*)(Q + qbase + (size_t)row * D_ + (ks + 1) * 32 + fcol);
                pk[i] = *(const float4*)(K + kbase + (size_t)row * D_ + (ks + 1) * 32 + fcol);
            }
        }

        Frag16 fa[2], fb[4];
        for (int mt = 0; mt < 2; ++mt) load_fragA(fa[mt], ldsQ[cur], wm + mt * 16, lane);
        for (int nt = 0; nt < 4; ++nt) load_fragB(fb[nt], ldsK[cur], wn + nt * 16, lane);

        for (int mt = 0; mt < 2; ++mt)
            for (int nt = 0; nt < 4; ++nt)
                acc[mt][nt] = __builtin_amdgcn_wmma_f32_16x16x32_bf16(
                    false, fa[mt].v, false, fb[nt].v,
                    (short)0, acc[mt][nt], false, false);

        if (ks + 1 < D_ / 32) {                // store next tile into the other buffer
            for (int i = 0; i < 4; ++i) {
                int row = frow + 32 * i;
                store_bf4(&ldsQ[cur ^ 1][row * LDS_STRIDE + fcol], pq[i]);
                store_bf4(&ldsK[cur ^ 1][row * LDS_STRIDE + fcol], pk[i]);
            }
        }
        __syncthreads();
    }

    // Write with causal mask (pre-scale, matching reference) then scale.
    const int half = lane >> 4, nl = lane & 15;
    for (int mt = 0; mt < 2; ++mt)
        for (int nt = 0; nt < 4; ++nt)
            for (int r = 0; r < 8; ++r) {
                int q = q0 + wm + mt * 16 + r + 8 * half;
                int k = k0 + wn + nt * 16 + nl;
                float s = acc[mt][nt][r];
                if (k > q) s -= 1e10f;
                S[((size_t)b * SQ_ + q) * SK_ + k] = s * INV_SCALE;
            }
}

// ---------------- Kernel 2: row softmax over causal prefix (padded to 128) ----------------
// grid = B*SQ, block = 256
__global__ __launch_bounds__(256) void softmax_kernel(float* __restrict__ S) {
    const int r = blockIdx.x;
    const int q = r & (SQ_ - 1);
    const int n = (((q >> 7) + 1) << 7);       // written prefix length
    float* __restrict__ row = S + (size_t)r * SK_;
    const int tid = threadIdx.x;

    __shared__ float sm[256];

    float v[8];
    int cnt = 0;
    float mx = -3.0e38f;
    for (int k = tid; k < n; k += 256) {
        float x = row[k];
        v[cnt++] = x;
        mx = fmaxf(mx, x);
    }
    sm[tid] = mx;
    __syncthreads();
    for (int s = 128; s > 0; s >>= 1) {
        if (tid < s) sm[tid] = fmaxf(sm[tid], sm[tid + s]);
        __syncthreads();
    }
    mx = sm[0];
    __syncthreads();

    float sum = 0.f;
    for (int i = 0; i < cnt; ++i) sum += __expf(v[i] - mx);
    sm[tid] = sum;
    __syncthreads();
    for (int s = 128; s > 0; s >>= 1) {
        if (tid < s) sm[tid] += sm[tid + s];
        __syncthreads();
    }
    const float inv = 1.f / sm[0];

    cnt = 0;
    for (int k = tid; k < n; k += 256)
        row[k] = __expf(v[cnt++] - mx) * inv;
}

// ---------------- Kernel 3: O = P @ V ----------------
// grid = (D/128, SQ/128, B), block = 256
__global__ __launch_bounds__(256) void pv_kernel(const float* __restrict__ S,
                                                 const float* __restrict__ V,
                                                 float* __restrict__ O) {
    const int dt = blockIdx.x, qt = blockIdx.y, b = blockIdx.z;
    const int q0 = qt * 128, d0 = dt * 128;

    __shared__ ushort_t ldsP[2][128 * LDS_STRIDE];   // [m][k]
    __shared__ ushort_t ldsV[2][128 * LDS_STRIDE];   // [n][k]  (V transposed)

    const int tid = threadIdx.x, lane = tid & 31, wid = tid >> 5;
    const int wm = (wid >> 1) * 32;
    const int wn = (wid & 1) * 64;

    const int prow = tid >> 3;                 // P fill: row base, col
    const int pcol = (tid & 7) * 4;
    const int vrow = tid & 31;                 // V fill: k row (0..31)
    const int vcol = (tid >> 5) * 4;           // V fill: d col base (0..28)

    v8f acc[2][4];
    for (int i = 0; i < 2; ++i)
        for (int j = 0; j < 4; ++j)
            acc[i][j] = {};

    const size_t pbase = ((size_t)b * SQ_ + q0) * SK_;
    const size_t vbase = ((size_t)b * SK_) * D_ + d0;

    const int nks = (qt + 1) * 4;              // causal: k < (qt+1)*128, step 32

    float4 pp[4], pv[4];
    for (int i = 0; i < 4; ++i) {
        pp[i] = *(const float4*)(S + pbase + (size_t)(prow + 32 * i) * SK_ + pcol);
        pv[i] = *(const float4*)(V + vbase + (size_t)vrow * D_ + vcol + 32 * i);
    }
    for (int i = 0; i < 4; ++i) {
        store_bf4(&ldsP[0][(prow + 32 * i) * LDS_STRIDE + pcol], pp[i]);
        int dc = vcol + 32 * i;
        ldsV[0][(dc + 0) * LDS_STRIDE + vrow] = f2bf1(pv[i].x);
        ldsV[0][(dc + 1) * LDS_STRIDE + vrow] = f2bf1(pv[i].y);
        ldsV[0][(dc + 2) * LDS_STRIDE + vrow] = f2bf1(pv[i].z);
        ldsV[0][(dc + 3) * LDS_STRIDE + vrow] = f2bf1(pv[i].w);
    }
    __syncthreads();

    for (int ks = 0; ks < nks; ++ks) {
        const int cur = ks & 1;
        if (ks + 1 < nks) {
            for (int i = 0; i < 4; ++i) {
                pp[i] = *(const float4*)(S + pbase + (size_t)(prow + 32 * i) * SK_ +
                                         (ks + 1) * 32 + pcol);
                pv[i] = *(const float4*)(V + vbase + (size_t)((ks + 1) * 32 + vrow) * D_ +
                                         vcol + 32 * i);
            }
        }

        Frag16 fa[2], fb[4];
        for (int mt = 0; mt < 2; ++mt) load_fragA(fa[mt], ldsP[cur], wm + mt * 16, lane);
        for (int nt = 0; nt < 4; ++nt) load_fragB(fb[nt], ldsV[cur], wn + nt * 16, lane);

        for (int mt = 0; mt < 2; ++mt)
            for (int nt = 0; nt < 4; ++nt)
                acc[mt][nt] = __builtin_amdgcn_wmma_f32_16x16x32_bf16(
                    false, fa[mt].v, false, fb[nt].v,
                    (short)0, acc[mt][nt], false, false);

        if (ks + 1 < nks) {
            for (int i = 0; i < 4; ++i) {
                store_bf4(&ldsP[cur ^ 1][(prow + 32 * i) * LDS_STRIDE + pcol], pp[i]);
                int dc = vcol + 32 * i;
                ldsV[cur ^ 1][(dc + 0) * LDS_STRIDE + vrow] = f2bf1(pv[i].x);
                ldsV[cur ^ 1][(dc + 1) * LDS_STRIDE + vrow] = f2bf1(pv[i].y);
                ldsV[cur ^ 1][(dc + 2) * LDS_STRIDE + vrow] = f2bf1(pv[i].z);
                ldsV[cur ^ 1][(dc + 3) * LDS_STRIDE + vrow] = f2bf1(pv[i].w);
            }
        }
        __syncthreads();
    }

    const int half = lane >> 4, nl = lane & 15;
    for (int mt = 0; mt < 2; ++mt)
        for (int nt = 0; nt < 4; ++nt)
            for (int r = 0; r < 8; ++r) {
                int q = q0 + wm + mt * 16 + r + 8 * half;
                int d = d0 + wn + nt * 16 + nl;
                O[((size_t)b * SQ_ + q) * D_ + d] = acc[mt][nt][r];
            }
}

extern "C" void kernel_launch(void* const* d_in, const int* in_sizes, int n_in,
                              void* d_out, int out_size, void* d_ws, size_t ws_size,
                              hipStream_t stream) {
    const float* Q = (const float*)d_in[0];
    const float* K = (const float*)d_in[1];
    const float* V = (const float*)d_in[2];
    float* S = (float*)d_ws;                    // B*SQ*SK fp32 = 64 MB scratch
    float* O = (float*)d_out;

    qk_kernel<<<dim3(SK_ / 128, SQ_ / 128, B_), 256, 0, stream>>>(Q, K, S);
    softmax_kernel<<<B_ * SQ_, 256, 0, stream>>>(S);
    pv_kernel<<<dim3(D_ / 128, SQ_ / 128, B_), 256, 0, stream>>>(S, V, O);
}
